// model_52261162058130
// MI455X (gfx1250) — matrix operand order repaired and checked by reading, not compiled, not run on hardware
//
#include <hip/hip_runtime.h>
#include <math.h>

typedef __attribute__((ext_vector_type(16))) _Float16 v16h;
typedef __attribute__((ext_vector_type(8)))  _Float16 v8h;
typedef __attribute__((ext_vector_type(8)))  float    v8f;

#define BB   4
#define N0   4096
#define NS   1024
#define KK   64
#define NBB  16
#define EPS_BN 1e-5f

// ---------------------------------------------------------------- FPS
__global__ void __launch_bounds__(256) fps_kernel(const float* __restrict__ pos,
                                                  int* __restrict__ idx) {
  int b = blockIdx.x;
  int t = threadIdx.x;
  const float* P = pos + (size_t)b * N0 * 3;
  float px[N0/256], py[N0/256], pz[N0/256], d[N0/256];
  #pragma unroll
  for (int j = 0; j < N0/256; ++j) {
    int p = t + j * 256;
    px[j] = P[p*3+0]; py[j] = P[p*3+1]; pz[j] = P[p*3+2];
    d[j] = 1e10f;
  }
  __shared__ float sval[256];
  __shared__ int   sid[256];
  __shared__ int   lastS;
  if (t == 0) { idx[b*NS] = 0; lastS = 0; }
  __syncthreads();
  for (int it = 1; it < NS; ++it) {
    int last = lastS;
    float lx = P[last*3+0], ly = P[last*3+1], lz = P[last*3+2];
    float best = -1.f; int bi = 0;
    #pragma unroll
    for (int j = 0; j < N0/256; ++j) {
      float dx = px[j]-lx, dy = py[j]-ly, dz = pz[j]-lz;
      float dd = dx*dx + dy*dy + dz*dz;
      float dm = fminf(d[j], dd);
      d[j] = dm;
      if (dm > best) { best = dm; bi = t + j*256; }
    }
    sval[t] = best; sid[t] = bi;
    __syncthreads();
    for (int s = 128; s > 0; s >>= 1) {
      if (t < s) {
        float v2 = sval[t+s]; int i2 = sid[t+s];
        if (v2 > sval[t] || (v2 == sval[t] && i2 < sid[t])) { sval[t] = v2; sid[t] = i2; }
      }
      __syncthreads();
    }
    if (t == 0) { lastS = sid[0]; idx[b*NS+it] = sid[0]; }
    __syncthreads();
  }
}

// -------------------------------------------------- gather pos -> f16 X (pad 32)
__global__ void gather_pos_kernel(const float* __restrict__ pos, const int* __restrict__ idx,
                                  _Float16* __restrict__ X) {
  int m = blockIdx.x * blockDim.x + threadIdx.x;
  if (m >= BB*NS) return;
  int b = m >> 10, i = m & (NS-1);
  int s = idx[b*NS + i];
  const float* p = pos + ((size_t)b*N0 + s) * 3;
  _Float16* x = X + (size_t)m * 32;
  x[0] = (_Float16)p[0]; x[1] = (_Float16)p[1]; x[2] = (_Float16)p[2];
  #pragma unroll
  for (int c = 3; c < 32; ++c) x[c] = (_Float16)0.f;
}

// -------------------------------------- pack weights into WMMA A-operand layout
// For row-tile rt (16 rows) and k-block kt (32 k), lane L holds 16 contiguous f16
// in contiguous k order (k = 16*(L/16) + j), the same order the GEMM loads B in.
__global__ void __launch_bounds__(512) pack_w_kernel(const float* __restrict__ W,
    _Float16* __restrict__ Wp, int Ci, int CpadIn) {
  int nkt = CpadIn >> 5;
  int rt = blockIdx.x / nkt, kt = blockIdx.x - rt*nkt;
  int L = threadIdx.x >> 4;     // lane 0..31
  int j = threadIdx.x & 15;     // element 0..15
  int half = L >> 4;
  int row = rt*16 + (L & 15);
  int v = j >> 1, e = j & 1;
  int kidx = half*16 + 2*v;
  int k = kt*32 + kidx + e;
  float w = (k < Ci) ? W[(size_t)row*Ci + k] : 0.f;
  Wp[(size_t)blockIdx.x*512 + threadIdx.x] = (_Float16)w;
}

// ---------------------------------------------------------------- WMMA GEMM
// X column-major f16 [M x CpadIn], Wp packed f16 (see pack_w_kernel),
// Y column-major [M x CpadOut].
template<int ITER, bool OUTF32>
__global__ void __launch_bounds__(128) gemm_wmma_kernel(
    const _Float16* __restrict__ Wp, const float* __restrict__ bias,
    const _Float16* __restrict__ X, void* __restrict__ Yout,
    int CpadIn, int CpadOut, int M)
{
  int lane = threadIdx.x & 31;
  int wave = threadIdx.x >> 5;
  int ln = lane & 15, half = lane >> 4;
  int row0 = blockIdx.y << 4;
  int colBase = (blockIdx.x * 4 + wave) * (16 * ITER);
  v8f zero = {};
  v8f acc[ITER];
  #pragma unroll
  for (int t = 0; t < ITER; ++t) acc[t] = zero;

  const _Float16* wp = Wp + ((size_t)blockIdx.y * (CpadIn >> 5)) * 512 + lane*16;
  size_t xoff[ITER];
  #pragma unroll
  for (int t = 0; t < ITER; ++t)
    xoff[t] = (size_t)(colBase + t*16 + ln) * CpadIn + half*16;

  for (int kb = 0; kb < CpadIn; kb += 32) {
    v16h a = *(const v16h*)wp;
    wp += 512;
    #pragma unroll
    for (int t = 0; t < ITER; ++t) {
      v16h bv = *(const v16h*)(X + xoff[t] + kb);   // direct GEP on kernarg -> global load
      acc[t] = __builtin_amdgcn_wmma_f32_16x16x32_f16(false, a, false, bv,
                                                      (short)0, acc[t], false, false);
    }
  }

  int orow = row0 + half*8;
  float bz[8];
  #pragma unroll
  for (int v = 0; v < 8; ++v) bz[v] = bias[orow + v];
  #pragma unroll
  for (int t = 0; t < ITER; ++t) {
    int col = colBase + t*16 + ln;
    if (OUTF32) {
      float* y = (float*)Yout + (size_t)col*CpadOut + orow;
      float4 lo = make_float4(acc[t][0]+bz[0], acc[t][1]+bz[1], acc[t][2]+bz[2], acc[t][3]+bz[3]);
      float4 hi = make_float4(acc[t][4]+bz[4], acc[t][5]+bz[5], acc[t][6]+bz[6], acc[t][7]+bz[7]);
      *(float4*)y = lo;
      *(float4*)(y + 4) = hi;
    } else {
      _Float16* y = (_Float16*)Yout + (size_t)col*CpadOut + orow;
      v8h h;
      #pragma unroll
      for (int v = 0; v < 8; ++v) h[v] = (_Float16)(acc[t][v] + bz[v]);
      *(v8h*)y = h;
    }
  }
}

// ---------------------------------------------------------------- BN stats & apply
__global__ void __launch_bounds__(256) bn_stats_kernel(const _Float16* __restrict__ Y,
    int C, int Cpad, int M, const float* __restrict__ g, const float* __restrict__ be,
    float* __restrict__ scale, float* __restrict__ shift) {
  int c = blockIdx.x;
  __shared__ float ss[256], sq[256];
  float s = 0.f, q = 0.f;
  for (int m = threadIdx.x; m < M; m += 256) {
    float v = (float)Y[(size_t)m*Cpad + c];
    s += v; q += v*v;
  }
  ss[threadIdx.x] = s; sq[threadIdx.x] = q;
  __syncthreads();
  for (int st = 128; st > 0; st >>= 1) {
    if (threadIdx.x < st) { ss[threadIdx.x] += ss[threadIdx.x+st]; sq[threadIdx.x] += sq[threadIdx.x+st]; }
    __syncthreads();
  }
  if (threadIdx.x == 0) {
    float mu  = ss[0] / (float)M;
    float var = sq[0] / (float)M - mu*mu;
    float sc  = g[c] * rsqrtf(var + EPS_BN);
    scale[c] = sc;
    shift[c] = be[c] - mu*sc;
  }
}

// vectorized: one v8h (8 channels) per thread; C is always a power of two >= 16
__global__ void bn_apply_kernel(_Float16* __restrict__ Y, int cgShift, int Cpad, int M,
    const float* __restrict__ scale, const float* __restrict__ shift, int relu) {
  int i = blockIdx.x * blockDim.x + threadIdx.x;
  if (i >= (M << cgShift)) return;
  int m = i >> cgShift;
  int cg = (i & ((1 << cgShift) - 1)) << 3;
  _Float16* y = Y + (size_t)m * Cpad + cg;
  v8h h = *(const v8h*)y;
  #pragma unroll
  for (int e = 0; e < 8; ++e) {
    float v = (float)h[e] * scale[cg + e] + shift[cg + e];
    if (relu) v = fmaxf(v, 0.f);
    h[e] = (_Float16)v;
  }
  *(v8h*)y = h;
}

// ---------------------------------------------------------------- max over points
__global__ void max_points_kernel(const _Float16* __restrict__ X, int Cpad, int C, int P,
                                  float* __restrict__ out) {
  int s = blockIdx.x;
  int c = threadIdx.x;
  if (c >= C) return;
  const _Float16* x = X + (size_t)s * P * Cpad + c;
  float m = -3.0e38f;
  for (int p = 0; p < P; ++p) m = fmaxf(m, (float)x[(size_t)p*Cpad]);
  out[(size_t)s*C + c] = m;
}

// ---------------------------------------------------------------- concat builders
__global__ void build_x21_kernel(const _Float16* __restrict__ S5, const float* __restrict__ w_pc,
    const int* __restrict__ sidx, const float* __restrict__ key_pts, _Float16* __restrict__ X) {
  int m = blockIdx.x * blockDim.x + threadIdx.x;
  if (m >= BB*KK*NS) return;
  int i = m & (NS-1); int bk = m >> 10; int k = bk & (KK-1); int b = bk >> 6;
  _Float16* x = X + (size_t)m * 96;
  const _Float16* s = S5 + ((size_t)(b<<10) + i) * 64;
  for (int c = 0; c < 64; ++c) x[c] = s[c];
  int sp = sidx[(b<<10) + i];
  x[64] = (_Float16)w_pc[((size_t)b*N0 + sp)*KK + k];
  const float* kp = key_pts + ((size_t)b*KK + k) * 3;
  x[65] = (_Float16)kp[0]; x[66] = (_Float16)kp[1]; x[67] = (_Float16)kp[2];
  for (int c = 68; c < 96; ++c) x[c] = (_Float16)0.f;
}

__global__ void build_x41_kernel(const float* __restrict__ key_fea,
    const _Float16* __restrict__ T5, _Float16* __restrict__ X) {
  int m = blockIdx.x * blockDim.x + threadIdx.x;
  if (m >= BB*KK*NS) return;
  int i = m & (NS-1); int bk = m >> 10; int b = bk >> 6;
  _Float16* x = X + (size_t)m * 128;
  const float* kf = key_fea + (size_t)bk * 64;
  for (int c = 0; c < 64; ++c) x[c] = (_Float16)kf[c];
  const _Float16* t = T5 + ((size_t)(b<<10) + i) * 128;
  for (int c = 0; c < 64; ++c) x[64+c] = t[c];
}

__global__ void build_x31_kernel(const float* __restrict__ key_fea,
    const float* __restrict__ key_pts, _Float16* __restrict__ X) {
  int m = blockIdx.x * blockDim.x + threadIdx.x;
  if (m >= BB*KK) return;
  _Float16* x = X + (size_t)m * 96;
  const float* kf = key_fea + (size_t)m * 64;
  for (int c = 0; c < 64; ++c) x[c] = (_Float16)kf[c];
  const float* kp = key_pts + (size_t)m * 3;
  x[64] = (_Float16)kp[0]; x[65] = (_Float16)kp[1]; x[66] = (_Float16)kp[2];
  for (int c = 67; c < 96; ++c) x[c] = (_Float16)0.f;
}

__global__ void build_x71_kernel(const float* __restrict__ key_fea,
    const float* __restrict__ key_pts, const float* __restrict__ basis,
    _Float16* __restrict__ X) {
  int m = blockIdx.x * blockDim.x + threadIdx.x;
  if (m >= BB*NBB*KK) return;
  int k = m & (KK-1); int bn = m >> 6; int b = bn >> 4;
  _Float16* x = X + (size_t)m * 96;
  const float* kf = key_fea + ((size_t)b*KK + k) * 64;
  for (int c = 0; c < 64; ++c) x[c] = (_Float16)kf[c];
  const float* kp = key_pts + ((size_t)b*KK + k) * 3;
  x[64] = (_Float16)kp[0]; x[65] = (_Float16)kp[1]; x[66] = (_Float16)kp[2];
  const float* bs = basis + (size_t)bn*192 + k*3;
  x[67] = (_Float16)bs[0]; x[68] = (_Float16)bs[1]; x[69] = (_Float16)bs[2];
  for (int c = 70; c < 96; ++c) x[c] = (_Float16)0.f;
}

__global__ void build_x51_kernel(const float* __restrict__ key_fea2,
    const float* __restrict__ basis, const float* __restrict__ key_pts,
    _Float16* __restrict__ X) {
  int m = blockIdx.x * blockDim.x + threadIdx.x;
  if (m >= BB*NBB*KK) return;
  int k = m & (KK-1); int bn = m >> 6; int b = bn >> 4;
  _Float16* x = X + (size_t)m * 160;
  const float* kf = key_fea2 + ((size_t)b*KK + k) * 128;
  for (int c = 0; c < 128; ++c) x[c] = (_Float16)kf[c];
  const float* bs = basis + (size_t)bn*192 + k*3;
  x[128] = (_Float16)bs[0]; x[129] = (_Float16)bs[1]; x[130] = (_Float16)bs[2];
  const float* kp = key_pts + ((size_t)b*KK + k) * 3;
  x[131] = (_Float16)kp[0]; x[132] = (_Float16)kp[1]; x[133] = (_Float16)kp[2];
  for (int c = 134; c < 160; ++c) x[c] = (_Float16)0.f;
}

__global__ void build_x61_kernel(const float* __restrict__ net5max,
    const float* __restrict__ cr, _Float16* __restrict__ X) {
  int m = threadIdx.x;
  if (m >= BB*NBB) return;
  _Float16* x = X + (size_t)m * 160;
  const float* f = net5max + (size_t)m * 128;
  for (int c = 0; c < 128; ++c) x[c] = (_Float16)f[c];
  x[128] = (_Float16)cr[m*2+0];
  x[129] = (_Float16)cr[m*2+1];
  for (int c = 130; c < 160; ++c) x[c] = (_Float16)0.f;
}

// ---------------------------------------------------------------- small GEMM (Co not /16)
__global__ void gemm_small_kernel(const float* __restrict__ W, const float* __restrict__ bias,
    const _Float16* __restrict__ X, float* __restrict__ Y, int Ci, int CpadIn, int Co, int M) {
  int i = blockIdx.x * blockDim.x + threadIdx.x;
  if (i >= Co*M) return;
  int m = i / Co, o = i - m*Co;
  const _Float16* x = X + (size_t)m * CpadIn;
  float s = bias[o];
  for (int c = 0; c < Ci; ++c) s += W[(size_t)o*Ci + c] * (float)x[c];
  Y[(size_t)m*Co + o] = s;
}

// ---------------------------------------------------------------- basis normalize
__global__ void __launch_bounds__(256) basis_kernel(const float* __restrict__ Y33,
                                                    float* __restrict__ basis) {
  int bb = blockIdx.x;                  // b*16 + nb
  int b = bb >> 4, nb = bb & 15;
  __shared__ float red[256];
  int j = threadIdx.x;
  float v = 0.f;
  if (j < 192) {
    int t = j*16 + nb;
    int k = t / 48, c = t - k*48;
    v = Y33[(size_t)((b<<6) + k)*48 + c];
  }
  red[j] = v*v;
  __syncthreads();
  for (int s = 128; s > 0; s >>= 1) { if (j < s) red[j] += red[j+s]; __syncthreads(); }
  float den = fmaxf(sqrtf(red[0]), 1e-6f);
  if (j < 192) basis[(size_t)bb*192 + j] = v / den;
}

// ---------------------------------------------------------------- coef range / coef
__global__ void coef_range_kernel(const float* __restrict__ Y73, float* __restrict__ cr) {
  int i = threadIdx.x;
  if (i >= 2*BB*NBB) return;
  int s = i >> 1, c = i & 1;
  float m = -3.0e38f;
  for (int k = 0; k < KK; ++k) m = fmaxf(m, Y73[((size_t)(s<<6) + k)*2 + c]);
  cr[i] = m * 0.1f * (c == 0 ? -1.f : 1.f);
}

__global__ void coef_kernel(const float* __restrict__ Y63, const float* __restrict__ cr,
                            float* __restrict__ coef) {
  int i = threadIdx.x;
  if (i >= BB*NBB) return;
  float c0 = 1.f / (1.f + expf(-Y63[i]));
  coef[i] = c0 * cr[i*2+0] + (1.f - c0) * cr[i*2+1];
}

// ---------------------------------------------------------------- def_key_pts / def_pc
__global__ void dkp_kernel(const float* __restrict__ key_pts, const float* __restrict__ coef,
                           const float* __restrict__ basis, float* __restrict__ out) {
  int i = blockIdx.x * blockDim.x + threadIdx.x;
  if (i >= BB*192) return;
  int b = i / 192; int j = i - b*192;
  float s = key_pts[i];
  for (int nb = 0; nb < NBB; ++nb)
    s += coef[b*NBB + nb] * basis[((size_t)b*NBB + nb)*192 + j];
  out[i] = s;
}

__global__ void __launch_bounds__(256) defpc_kernel(const float* __restrict__ w_pc,
    const float* __restrict__ dkp, float* __restrict__ out) {
  __shared__ float s[192];
  int b = blockIdx.x;
  int i = blockIdx.y * 256 + threadIdx.x;
  if (threadIdx.x < 192) s[threadIdx.x] = dkp[b*192 + threadIdx.x];
  __syncthreads();
  const float* w = w_pc + ((size_t)b*N0 + i) * KK;
  float a0 = 0.f, a1 = 0.f, a2 = 0.f;
  for (int k = 0; k < KK; ++k) {
    float wv = w[k];
    a0 += wv * s[k*3+0]; a1 += wv * s[k*3+1]; a2 += wv * s[k*3+2];
  }
  float* o = out + ((size_t)b*N0 + i) * 3;
  o[0] = a0; o[1] = a1; o[2] = a2;
}

// ---------------------------------------------------------------- chamfer
__global__ void __launch_bounds__(256) chamfer_min_kernel(const float* __restrict__ A,
    const float* __restrict__ Bp, float* __restrict__ partial, int partOff) {
  int b = blockIdx.x;
  int i = blockIdx.y * 256 + threadIdx.x;
  const float* a = A + ((size_t)b*N0 + i) * 3;
  float ax = a[0], ay = a[1], az = a[2];
  float mn = 3.0e38f;
  __shared__ float tile[768];
  for (int j0 = 0; j0 < N0; j0 += 256) {
    const float* bp = Bp + ((size_t)b*N0 + j0) * 3;
    __syncthreads();
    for (int t = threadIdx.x; t < 768; t += 256) tile[t] = bp[t];
    __syncthreads();
    for (int j = 0; j < 256; ++j) {
      float dx = ax - tile[j*3+0], dy = ay - tile[j*3+1], dz = az - tile[j*3+2];
      mn = fminf(mn, dx*dx + dy*dy + dz*dz);
    }
  }
  __shared__ float red[256];
  red[threadIdx.x] = mn;
  __syncthreads();
  for (int s = 128; s > 0; s >>= 1) { if (threadIdx.x < s) red[threadIdx.x] += red[threadIdx.x+s]; __syncthreads(); }
  if (threadIdx.x == 0) partial[partOff + b*16 + blockIdx.y] = red[0];
}

__global__ void cham_final_kernel(const float* __restrict__ partial, float* __restrict__ loss) {
  __shared__ float red[128];
  int t = threadIdx.x;
  red[t] = partial[t];
  __syncthreads();
  for (int s = 64; s > 0; s >>= 1) { if (t < s) red[t] += red[t+s]; __syncthreads(); }
  if (t == 0) loss[0] = red[0] / (float)(BB*N0);
}

// ================================================================ host
enum {
  L_pn1 = 0, L_pn2, L_tpn1, L_tpn2, L_conv11, L_conv12, L_conv13,
  L_tconv11, L_tconv12, L_tconv13, L_conv21, L_conv22,
  L_conv31, L_conv32, L_conv33, L_conv41, L_conv42, L_conv43,
  L_conv51, L_conv52, L_conv53, L_conv61, L_conv62, L_conv63,
  L_conv71, L_conv72, L_conv73
};

extern "C" void kernel_launch(void* const* d_in, const int* in_sizes, int n_in,
                              void* d_out, int out_size, void* d_ws, size_t ws_size,
                              hipStream_t stream) {
  (void)in_sizes; (void)n_in; (void)out_size; (void)ws_size;
  const float* src_pc  = (const float*)d_in[0];
  const float* tar_pc  = (const float*)d_in[1];
  const float* key_pts = (const float*)d_in[2];
  const float* w_pc    = (const float*)d_in[3];
  auto PW = [&](int L){ return (const float*)d_in[4 + 4*L + 0]; };
  auto PB = [&](int L){ return (const float*)d_in[4 + 4*L + 1]; };
  auto PG = [&](int L){ return (const float*)d_in[4 + 4*L + 2]; };
  auto PE = [&](int L){ return (const float*)d_in[4 + 4*L + 3]; };

  char* base = (char*)d_ws;
  size_t off = 0;
  auto alloc = [&](size_t bytes) -> void* {
    void* p = base + off;
    off += (bytes + 255) & ~(size_t)255;
    return p;
  };

  const size_t M1 = (size_t)BB*NS;        // 4096
  const size_t M2 = (size_t)BB*KK*NS;     // 262144
  const size_t M3 = (size_t)BB*KK;        // 256
  const size_t M4 = (size_t)BB*NBB*KK;    // 4096
  const size_t M5 = (size_t)BB*NBB;       // 64

  int*       sidxS   = (int*)alloc(M1*4);
  int*       sidxT   = (int*)alloc(M1*4);
  _Float16*  S0 = (_Float16*)alloc(M1*32*2);
  _Float16*  S1 = (_Float16*)alloc(M1*64*2);
  _Float16*  S2 = (_Float16*)alloc(M1*128*2);
  _Float16*  S3 = (_Float16*)alloc(M1*128*2);
  _Float16*  S4 = (_Float16*)alloc(M1*64*2);
  _Float16*  S5 = (_Float16*)alloc(M1*64*2);
  _Float16*  T0 = (_Float16*)alloc(M1*32*2);
  _Float16*  T1 = (_Float16*)alloc(M1*64*2);
  _Float16*  T2 = (_Float16*)alloc(M1*128*2);
  _Float16*  T3 = (_Float16*)alloc(M1*128*2);
  _Float16*  T4 = (_Float16*)alloc(M1*128*2);
  _Float16*  T5 = (_Float16*)alloc(M1*128*2);
  _Float16*  BIGX = (_Float16*)alloc(M2*256*2);   // two 128-ch halves
  _Float16*  BIGY = (_Float16*)alloc(M2*256*2);
  _Float16*  BIGXlo = BIGX;
  _Float16*  BIGXhi = BIGX + (size_t)128*M2;
  float*     key_fea  = (float*)alloc(M3*64*4);
  float*     key_fea2 = (float*)alloc(M3*128*4);
  _Float16*  X31 = (_Float16*)alloc(M3*96*2);
  _Float16*  Y31 = (_Float16*)alloc(M3*256*2);
  _Float16*  Y32 = (_Float16*)alloc(M3*512*2);
  float*     Y33 = (float*)alloc(M3*48*4);
  float*     basis = (float*)alloc(M5*192*4);
  _Float16*  X71 = (_Float16*)alloc(M4*96*2);
  _Float16*  Y71 = (_Float16*)alloc(M4*32*2);
  _Float16*  Y72 = (_Float16*)alloc(M4*32*2);
  float*     Y73 = (float*)alloc(M4*2*4);
  float*     cr  = (float*)alloc(M5*2*4);
  _Float16*  X51 = (_Float16*)alloc(M4*160*2);
  _Float16*  Y51 = (_Float16*)alloc(M4*256*2);
  _Float16*  Y52 = (_Float16*)alloc(M4*128*2);
  _Float16*  Y53 = (_Float16*)alloc(M4*128*2);
  float*     net5max = (float*)alloc(M5*128*4);
  _Float16*  X61 = (_Float16*)alloc(M5*160*2);
  _Float16*  Y61 = (_Float16*)alloc(M5*64*2);
  _Float16*  Y62 = (_Float16*)alloc(M5*32*2);
  float*     Y63 = (float*)alloc(M5*4);
  float*     coef = (float*)alloc(M5*4);
  float*     scaleB = (float*)alloc(512*4);
  float*     shiftB = (float*)alloc(512*4);
  float*     partial = (float*)alloc(128*4);
  _Float16*  wpack = (_Float16*)alloc((size_t)512*512*2);  // packed weights (reused per layer)

  float* out_dkp  = (float*)d_out;            // 768
  float* out_dpc  = (float*)d_out + 768;      // 49152
  float* out_loss = (float*)d_out + 49920;    // 1

  auto bn_layer = [&](int L, const _Float16* Xin, int Ci, int CpadIn,
                      _Float16* Yo, int Co, int CpadOut, int M, bool relu, int iter) {
    pack_w_kernel<<<(Co/16)*(CpadIn/32),512,0,stream>>>(PW(L), wpack, Ci, CpadIn);
    dim3 g(M / (64 * iter), Co / 16);
    if (iter == 4)
      gemm_wmma_kernel<4,false><<<g,128,0,stream>>>(wpack,PB(L),Xin,Yo,CpadIn,CpadOut,M);
    else
      gemm_wmma_kernel<1,false><<<g,128,0,stream>>>(wpack,PB(L),Xin,Yo,CpadIn,CpadOut,M);
    bn_stats_kernel<<<Co,256,0,stream>>>(Yo,Co,CpadOut,M,PG(L),PE(L),scaleB,shiftB);
    int cgShift = __builtin_ctz((unsigned)Co) - 3;   // log2(C/8); C is 16..512 pow2
    int total = M << cgShift;
    bn_apply_kernel<<<(total+255)/256,256,0,stream>>>(Yo,cgShift,CpadOut,M,scaleB,shiftB,relu?1:0);
  };

  // 1. FPS + gather
  fps_kernel<<<BB,256,0,stream>>>(src_pc, sidxS);
  fps_kernel<<<BB,256,0,stream>>>(tar_pc, sidxT);
  gather_pos_kernel<<<(int)((M1+255)/256),256,0,stream>>>(src_pc, sidxS, S0);
  gather_pos_kernel<<<(int)((M1+255)/256),256,0,stream>>>(tar_pc, sidxT, T0);

  // 2. src / tar encoder chains
  bn_layer(L_pn1,    S0,   3,  32, S1,  64,  64, (int)M1, true, 4);
  bn_layer(L_pn2,    S1,  64,  64, S2, 128, 128, (int)M1, true, 4);
  bn_layer(L_conv11, S2, 128, 128, S3, 128, 128, (int)M1, true, 4);
  bn_layer(L_conv12, S3, 128, 128, S4,  64,  64, (int)M1, true, 4);
  bn_layer(L_conv13, S4,  64,  64, S5,  64,  64, (int)M1, true, 4);
  bn_layer(L_tpn1,    T0,   3,  32, T1,  64,  64, (int)M1, true, 4);
  bn_layer(L_tpn2,    T1,  64,  64, T2, 128, 128, (int)M1, true, 4);
  bn_layer(L_tconv11, T2, 128, 128, T3, 128, 128, (int)M1, true, 4);
  bn_layer(L_tconv12, T3, 128, 128, T4, 128, 128, (int)M1, true, 4);
  bn_layer(L_tconv13, T4, 128, 128, T5, 128, 128, (int)M1, true, 4);

  // 3. conv21/22 -> key_fea
  build_x21_kernel<<<(int)((M2+255)/256),256,0,stream>>>(S5, w_pc, sidxS, key_pts, BIGXlo);
  bn_layer(L_conv21, BIGXlo, 68, 96, BIGY,   64, 64, (int)M2, true,  4);
  bn_layer(L_conv22, BIGY,   64, 64, BIGXhi, 64, 64, (int)M2, false, 4);
  max_points_kernel<<<(int)M3,64,0,stream>>>(BIGXhi, 64, 64, NS, key_fea);

  // 4. conv31/32/33 -> basis
  build_x31_kernel<<<1,256,0,stream>>>(key_fea, key_pts, X31);
  bn_layer(L_conv31, X31,  67,  96, Y31, 256, 256, (int)M3, true, 1);
  bn_layer(L_conv32, Y31, 256, 256, Y32, 512, 512, (int)M3, true, 1);
  {
    pack_w_kernel<<<(48/16)*(512/32),512,0,stream>>>(PW(L_conv33), wpack, 512, 512);
    dim3 g((int)M3/64, 48/16);
    gemm_wmma_kernel<1,true><<<g,128,0,stream>>>(wpack,PB(L_conv33),Y32,Y33,512,48,(int)M3);
  }
  basis_kernel<<<(int)M5,256,0,stream>>>(Y33, basis);

  // 5. conv71/72/73 -> coef_range
  build_x71_kernel<<<(int)((M4+255)/256),256,0,stream>>>(key_fea, key_pts, basis, X71);
  bn_layer(L_conv71, X71, 70, 96, Y71, 32, 32, (int)M4, true, 4);
  bn_layer(L_conv72, Y71, 32, 32, Y72, 16, 32, (int)M4, true, 4);
  gemm_small_kernel<<<(int)((2*M4+255)/256),256,0,stream>>>(PW(L_conv73),PB(L_conv73),Y72,Y73,16,32,2,(int)M4);
  coef_range_kernel<<<1,128,0,stream>>>(Y73, cr);

  // 6. conv41/42/43 -> key_fea2
  build_x41_kernel<<<(int)((M2+255)/256),256,0,stream>>>(key_fea, T5, BIGXlo);
  bn_layer(L_conv41, BIGXlo, 128, 128, BIGY,   256, 256, (int)M2, true, 4);
  bn_layer(L_conv42, BIGY,   256, 256, BIGXhi, 128, 128, (int)M2, true, 4);
  bn_layer(L_conv43, BIGXhi, 128, 128, BIGXlo, 128, 128, (int)M2, true, 4);
  max_points_kernel<<<(int)M3,128,0,stream>>>(BIGXlo, 128, 128, NS, key_fea2);

  // 7. conv51/52/53 -> net5max
  build_x51_kernel<<<(int)((M4+255)/256),256,0,stream>>>(key_fea2, basis, key_pts, X51);
  bn_layer(L_conv51, X51, 134, 160, Y51, 256, 256, (int)M4, true,  4);
  bn_layer(L_conv52, Y51, 256, 256, Y52, 128, 128, (int)M4, true,  4);
  bn_layer(L_conv53, Y52, 128, 128, Y53, 128, 128, (int)M4, false, 4);
  max_points_kernel<<<(int)M5,128,0,stream>>>(Y53, 128, 128, KK, net5max);

  // 8. conv61/62/63 -> coef
  build_x61_kernel<<<1,64,0,stream>>>(net5max, cr, X61);
  bn_layer(L_conv61, X61, 130, 160, Y61, 64, 64, (int)M5, true, 1);
  bn_layer(L_conv62, Y61,  64,  64, Y62, 32, 32, (int)M5, true, 1);
  gemm_small_kernel<<<1,256,0,stream>>>(PW(L_conv63),PB(L_conv63),Y62,Y63,32,32,1,(int)M5);
  coef_kernel<<<1,64,0,stream>>>(Y63, cr, coef);

  // 9. outputs
  dkp_kernel<<<3,256,0,stream>>>(key_pts, coef, basis, out_dkp);
  defpc_kernel<<<dim3(BB, N0/256),256,0,stream>>>(w_pc, out_dkp, out_dpc);
  chamfer_min_kernel<<<dim3(BB, N0/256),256,0,stream>>>(out_dpc, tar_pc, partial, 0);
  chamfer_min_kernel<<<dim3(BB, N0/256),256,0,stream>>>(tar_pc, out_dpc, partial, 64);
  cham_final_kernel<<<1,128,0,stream>>>(partial, out_loss);
}
